// Qwen3Attention_74491912781949
// MI455X (gfx1250) — compile-verified, matrix operand order checked by reading, not run on hardware
//
#include <hip/hip_runtime.h>
#include <math.h>

// ---------------------------------------------------------------------------
// CDNA5 (gfx1250) Qwen3 attention block, bf16 WMMA pipeline.
// wave32; v_wmma_f32_16x16x32_bf16 for all matmuls; TDM (tensor_load_to_lds)
// for the contiguous K-tile staging in flash attention.
// ---------------------------------------------------------------------------

typedef __attribute__((ext_vector_type(16))) __bf16 v16bf;
typedef __attribute__((ext_vector_type(8)))  float  v8f;
typedef __attribute__((ext_vector_type(4)))  unsigned int u32x4;
typedef __attribute__((ext_vector_type(8)))  unsigned int u32x8;

#define BQ   2
#define SQ   2048
#define HIDQ 2048
#define HQ   16
#define HKVQ 8
#define DQ   128
#define QSCALE 0.08838834764831845f   // 1/sqrt(128)

__device__ __forceinline__ v8f wmma_bf16(v16bf a, v16bf b, v8f c) {
    return __builtin_amdgcn_wmma_f32_16x16x32_bf16(
        /*neg_a=*/false, a, /*neg_b=*/false, b,
        /*c_mod=*/(short)0, c, /*reuse_a=*/false, /*reuse_b=*/false);
}

// ---------------------------------------------------------------------------
// Generic bf16-WMMA GEMM: C[M,N] = A[M,K] * W[K,N]
//   A: fp32 (converted during LDS stage) or bf16; W: fp32; C: fp32 or bf16
// Tile 128x128, K-step 32, double-buffered LDS (one barrier / K-step).
// 256 threads = 8 waves (4 M x 2 N), each wave computes 32x64.
// ---------------------------------------------------------------------------
template <bool A_F32, bool OUT_F32>
__global__ __launch_bounds__(256) void gemm_wmma_bf16(
    const void* __restrict__ Ap, const float* __restrict__ W,
    void* __restrict__ Cp, int M, int N, int K)
{
    __shared__ __align__(64) __bf16 aT[2][128 * 32];   // [m][k]
    __shared__ __align__(64) __bf16 bT[2][128 * 32];   // [n][k]  (W transposed)

    const int tid  = threadIdx.x;
    const int lane = tid & 31;
    const int wid  = tid >> 5;
    const int wm   = wid & 3;     // M offset wm*32
    const int wn   = wid >> 2;    // N offset wn*64
    const int lr   = lane & 15;
    const int hi   = lane >> 4;

    const int mb = blockIdx.y * 128;
    const int nb = blockIdx.x * 128;

    const float*  Af = (const float*)Ap;
    const __bf16* Ab = (const __bf16*)Ap;

    auto stage = [&](int kk, int p) {
        // A tile [128][32]
        #pragma unroll
        for (int r = 0; r < 4; ++r) {
            const int row = r * 32 + (tid >> 3);
            const int k4  = (tid & 7) * 4;
            __bf16* dst = &aT[p][row * 32 + k4];
            if constexpr (A_F32) {
                const float4 v = *(const float4*)(Af + (size_t)(mb + row) * K + kk + k4);
                dst[0] = (__bf16)v.x; dst[1] = (__bf16)v.y;
                dst[2] = (__bf16)v.z; dst[3] = (__bf16)v.w;
            } else {
                *(uint2*)dst = *(const uint2*)(Ab + (size_t)(mb + row) * K + kk + k4);
            }
        }
        // W tile as [n][k]
        #pragma unroll
        for (int r = 0; r < 4; ++r) {
            const int k  = r * 8 + (tid >> 5);
            const int n4 = (tid & 31) * 4;
            const float4 v = *(const float4*)(W + (size_t)(kk + k) * N + nb + n4);
            bT[p][(n4 + 0) * 32 + k] = (__bf16)v.x;
            bT[p][(n4 + 1) * 32 + k] = (__bf16)v.y;
            bT[p][(n4 + 2) * 32 + k] = (__bf16)v.z;
            bT[p][(n4 + 3) * 32 + k] = (__bf16)v.w;
        }
    };

    v8f acc[2][4] = {};

    stage(0, 0);
    __syncthreads();

    for (int kk = 0; kk < K; kk += 32) {
        const int p = (kk >> 5) & 1;
        if (kk + 32 < K) stage(kk + 32, p ^ 1);   // overlap with WMMA below

        v16bf af[2], bf[4];
        #pragma unroll
        for (int mf = 0; mf < 2; ++mf)
            af[mf] = *(const v16bf*)&aT[p][(wm * 32 + mf * 16 + lr) * 32 + hi * 16];
        #pragma unroll
        for (int nf = 0; nf < 4; ++nf)
            bf[nf] = *(const v16bf*)&bT[p][(wn * 64 + nf * 16 + lr) * 32 + hi * 16];

        #pragma unroll
        for (int mf = 0; mf < 2; ++mf)
            #pragma unroll
            for (int nf = 0; nf < 4; ++nf)
                acc[mf][nf] = wmma_bf16(af[mf], bf[nf], acc[mf][nf]);

        __syncthreads();
    }

    // epilogue: C-fragment layout -> global
    #pragma unroll
    for (int mf = 0; mf < 2; ++mf) {
        #pragma unroll
        for (int nf = 0; nf < 4; ++nf) {
            #pragma unroll
            for (int r = 0; r < 8; ++r) {
                const int row = mb + wm * 32 + mf * 16 + r + 8 * hi;
                const int col = nb + wn * 64 + nf * 16 + lr;
                const float v = acc[mf][nf][r];
                if constexpr (OUT_F32)
                    ((float*)Cp)[(size_t)row * N + col] = v;
                else
                    ((__bf16*)Cp)[(size_t)row * N + col] = (__bf16)v;
            }
        }
    }
}

// ---------------------------------------------------------------------------
// RoPE + repack: qkv bf16 [B*S, 4096] ->
//   Q bf16 [B,H,S,D] (rotary + 1/sqrt(D)), K bf16 [B,HKV,S,D] (rotary),
//   V bf16 [B,HKV,S,D]
// ---------------------------------------------------------------------------
__global__ __launch_bounds__(256) void rope_pack(
    const __bf16* __restrict__ qkv,
    __bf16* __restrict__ Qb, __bf16* __restrict__ Kb, __bf16* __restrict__ Vb)
{
    const int idx  = blockIdx.x * blockDim.x + threadIdx.x;
    const int i    = idx & 63;
    const int head = (idx >> 6) & 31;
    const int bs   = idx >> 11;          // 0..B*S-1
    const int s    = bs & (SQ - 1);
    const int b    = bs >> 11;

    const __bf16* src = qkv + (size_t)bs * 4096;

    if (head < HQ) {
        const float x1 = (float)src[head * DQ + i];
        const float x2 = (float)src[head * DQ + i + 64];
        const float ang = (float)s * __powf(10000.0f, -(float)i * (1.0f / 64.0f));
        const float c = cosf(ang), sn = sinf(ang);
        __bf16* dst = Qb + ((size_t)(b * HQ + head) * SQ + s) * DQ;
        dst[i]      = (__bf16)((x1 * c - x2 * sn) * QSCALE);
        dst[i + 64] = (__bf16)((x2 * c + x1 * sn) * QSCALE);
    } else if (head < HQ + HKVQ) {
        const int hk = head - HQ;
        const float x1 = (float)src[HQ * DQ + hk * DQ + i];
        const float x2 = (float)src[HQ * DQ + hk * DQ + i + 64];
        const float ang = (float)s * __powf(10000.0f, -(float)i * (1.0f / 64.0f));
        const float c = cosf(ang), sn = sinf(ang);
        __bf16* dst = Kb + ((size_t)(b * HKVQ + hk) * SQ + s) * DQ;
        dst[i]      = (__bf16)(x1 * c - x2 * sn);
        dst[i + 64] = (__bf16)(x2 * c + x1 * sn);
    } else {
        const int hv = head - (HQ + HKVQ);
        __bf16* dst = Vb + ((size_t)(b * HKVQ + hv) * SQ + s) * DQ;
        dst[i]      = src[(HQ + HKVQ) * DQ + hv * DQ + i];
        dst[i + 64] = src[(HQ + HKVQ) * DQ + hv * DQ + i + 64];
    }
}

// ---------------------------------------------------------------------------
// Flash attention (causal, GQA 2:1). Grid (S/64, B*H), block = 128 (4 waves).
// K tile (contiguous 32x128 bf16 = 8 KB) staged by the Tensor Data Mover:
// wave 0 issues tensor_load_to_lds with a D# built in SGPRs, waits
// TENSORcnt==0, then the block barrier publishes the tile.
// ---------------------------------------------------------------------------
__global__ __launch_bounds__(128) void flash_attn(
    const __bf16* __restrict__ Qb, const __bf16* __restrict__ Kb,
    const __bf16* __restrict__ Vb, __bf16* __restrict__ attn)
{
    __shared__ __align__(64) __bf16 klds[32 * 128];     // [kv][d]  (TDM target)
    __shared__ __align__(64) __bf16 vt[128 * 32];       // [d][kv]  (V transposed)
    __shared__ __align__(64) __bf16 plds[4][16 * 32];   // per-wave P scratch

    const int tid  = threadIdx.x;
    const int w    = tid >> 5;
    const int lane = tid & 31;
    const int lr   = lane & 15;
    const int hi   = lane >> 4;

    const int bh  = blockIdx.y;
    const int b   = bh >> 4;
    const int h   = bh & 15;
    const int hkv = h >> 1;               // GQA: group size 2
    const int q0  = blockIdx.x * 64;
    const int qw  = q0 + w * 16;          // this wave's first query row

    // Q fragments (A layout): lane lr = row, hi selects K-half; 4 chunks of 32
    v16bf qf[4];
    const __bf16* qptr =
        Qb + ((size_t)(b * HQ + h) * SQ + (qw + lr)) * DQ + hi * 16;
    #pragma unroll
    for (int c = 0; c < 4; ++c) qf[c] = *(const v16bf*)(qptr + c * 32);

    float row_m[8], row_l[8];
    #pragma unroll
    for (int r = 0; r < 8; ++r) { row_m[r] = -1e30f; row_l[r] = 0.0f; }
    v8f o[8] = {};

    const __bf16* Kg = Kb + (size_t)(b * HKVQ + hkv) * SQ * DQ;
    const __bf16* Vg = Vb + (size_t)(b * HKVQ + hkv) * SQ * DQ;

    const unsigned int klds_lo = (unsigned int)(size_t)(&klds[0]); // LDS byte addr

    const int ntiles = (q0 + 64) >> 5;    // causal: kv < q0+64

    for (int kt = 0; kt < ntiles; ++kt) {
        const int kv0 = kt * 32;

        // ---- K tile via TDM: contiguous 8192-byte 1-D transfer ----
        if (w == 0) {
            const unsigned long long ga =
                (unsigned long long)(size_t)(Kg + (size_t)kv0 * DQ);
            u32x4 g0;
            g0[0] = 1u;                                   // count=1, user D#
            g0[1] = klds_lo;                              // lds_addr (bytes)
            g0[2] = (unsigned int)(ga & 0xffffffffu);     // global_addr[31:0]
            g0[3] = (unsigned int)((ga >> 32) & 0x01ffffffu) | (2u << 30); // [56:32]|type=2
            u32x8 g1;
            g1[0] = 0x00010000u;        // workgroup_mask=0, data_size=1 (2 bytes)
            g1[1] = (4096u << 16);      // tensor_dim0[15:0] -> bits[63:48]
            g1[2] = (1u << 16);         // tensor_dim0 hi=0, tensor_dim1=1
            g1[3] = (4096u << 16);      // tile_dim0 = 4096 elements (8 KB)
            g1[4] = 0u;                 // tile_dim1=0 (unused), tile_dim2=0
            g1[5] = 4096u;              // tensor_dim0_stride lo
            g1[6] = 0u;                 // stride hi, tensor_dim1_stride lo
            g1[7] = 0u;
            asm volatile("tensor_load_to_lds %0, %1" :: "s"(g0), "s"(g1) : "memory");
        }

        // ---- V^T tile staged by all threads ----
        #pragma unroll
        for (int r = 0; r < 8; ++r) {
            const int row = r * 4 + (tid >> 5);
            const int d0  = (tid & 31) * 4;
            const __bf16* vs = Vg + (size_t)(kv0 + row) * DQ + d0;
            vt[(d0 + 0) * 32 + row] = vs[0];
            vt[(d0 + 1) * 32 + row] = vs[1];
            vt[(d0 + 2) * 32 + row] = vs[2];
            vt[(d0 + 3) * 32 + row] = vs[3];
        }

        if (w == 0) __builtin_amdgcn_s_wait_tensorcnt(0);
        __syncthreads();

        // ---- scores: two 16x16 fragments over 32 kv columns ----
        v8f sc[2] = {};
        #pragma unroll
        for (int c = 0; c < 4; ++c) {
            #pragma unroll
            for (int nf = 0; nf < 2; ++nf) {
                v16bf bk = *(const v16bf*)
                    &klds[(nf * 16 + lr) * 128 + c * 32 + hi * 16];
                sc[nf] = wmma_bf16(qf[c], bk, sc[nf]);
            }
        }

        // ---- causal mask + online softmax (xor-shuffles stay in 16-lane halves) ----
        #pragma unroll
        for (int r = 0; r < 8; ++r) {
            const int qg = qw + r + 8 * hi;
            float s0 = sc[0][r]; if (kv0 + lr      > qg) s0 = -1e30f;
            float s1 = sc[1][r]; if (kv0 + 16 + lr > qg) s1 = -1e30f;

            float mx = fmaxf(s0, s1);
            #pragma unroll
            for (int m = 1; m < 16; m <<= 1)
                mx = fmaxf(mx, __shfl_xor(mx, m, 32));
            const float nm    = fmaxf(row_m[r], mx);
            const float alpha = __expf(row_m[r] - nm);
            const float p0 = __expf(s0 - nm);
            const float p1 = __expf(s1 - nm);
            float rs = p0 + p1;
            #pragma unroll
            for (int m = 1; m < 16; m <<= 1)
                rs += __shfl_xor(rs, m, 32);
            row_l[r] = row_l[r] * alpha + rs;
            row_m[r] = nm;
            #pragma unroll
            for (int df = 0; df < 8; ++df) o[df][r] *= alpha;

            // write P (C layout -> LDS row-major for A-fragment reload)
            __bf16* pw = &plds[w][(r + 8 * hi) * 32];
            pw[lr]      = (__bf16)p0;
            pw[16 + lr] = (__bf16)p1;
        }

        // in-wave LDS RAW: make stores visible before re-reading
        asm volatile("s_wait_dscnt 0x0" ::: "memory");
        __builtin_amdgcn_wave_barrier();

        v16bf aP = *(const v16bf*)&plds[w][lr * 32 + hi * 16];

        // ---- PV: P(16x32) x V(32x128) ----
        #pragma unroll
        for (int df = 0; df < 8; ++df) {
            v16bf bv = *(const v16bf*)&vt[(df * 16 + lr) * 32 + hi * 16];
            o[df] = wmma_bf16(aP, bv, o[df]);
        }
        __syncthreads();
    }

    // ---- normalize + store bf16 [B*S, H*D] ----
    #pragma unroll
    for (int r = 0; r < 8; ++r) {
        const float inv = 1.0f / row_l[r];
        const int srow  = qw + r + 8 * hi;
        __bf16* orow = attn + ((size_t)(b * SQ + srow)) * (HQ * DQ) + h * DQ;
        #pragma unroll
        for (int df = 0; df < 8; ++df)
            orow[df * 16 + lr] = (__bf16)(o[df][r] * inv);
    }
}

// ---------------------------------------------------------------------------
extern "C" void kernel_launch(void* const* d_in, const int* in_sizes, int n_in,
                              void* d_out, int out_size, void* d_ws, size_t ws_size,
                              hipStream_t stream)
{
    const float* hidden = (const float*)d_in[0];
    // d_in[1] = positions (arange(S)); computed analytically in rope_pack
    const float* w_qkv  = (const float*)d_in[2];
    const float* w_o    = (const float*)d_in[3];

    char* ws = (char*)d_ws;
    // region A (32 MB): qkv bf16 [4096,4096]; later aliased by attn bf16 [4096,2048]
    __bf16* qkv  = (__bf16*)ws;
    __bf16* attn = (__bf16*)ws;
    // region B (32 MB): Q/K/V bf16
    __bf16* Qb = (__bf16*)(ws + (size_t)33554432);
    __bf16* Kb = (__bf16*)(ws + (size_t)33554432 + 16777216);
    __bf16* Vb = (__bf16*)(ws + (size_t)33554432 + 16777216 + 8388608);

    // 1) QKV projection: [4096,2048] x [2048,4096] -> bf16 qkv
    gemm_wmma_bf16<true, false><<<dim3(32, 32), 256, 0, stream>>>(
        hidden, w_qkv, qkv, 4096, 4096, 2048);

    // 2) RoPE + repack into Q/K/V (bf16, WMMA-friendly layouts)
    rope_pack<<<32768, 256, 0, stream>>>(qkv, Qb, Kb, Vb);

    // 3) Causal GQA flash attention -> attn bf16 [4096, 2048]
    flash_attn<<<dim3(32, 32), 128, 0, stream>>>(Qb, Kb, Vb, attn);

    // 4) Output projection: [4096,2048] x [2048,2048] -> fp32 d_out
    gemm_wmma_bf16<false, true><<<dim3(16, 32), 256, 0, stream>>>(
        attn, w_o, d_out, 4096, 2048, 2048);
}
